// GCN_9225589752224
// MI455X (gfx1250) — compile-verified, matrix operand order checked by reading, not run on hardware
//
#include <hip/hip_runtime.h>
#include <hip/hip_bf16.h>
#include <cstdint>
#include <cstddef>

#define N_NODES  50000
#define N_EDGES  500000
#define N_GRAPHS 256
#define D_IN     768
#define D_H1     256
#define D_H2     128
#define D_H3     64
#define D_OUT    2

#define LDS_STR  48   // halves per LDS tile row (32 data + 16 pad, keeps 16B alignment)

typedef __attribute__((ext_vector_type(16))) __bf16 v16bf;
typedef __attribute__((ext_vector_type(8)))  float  v8f;

// float -> bf16 round-to-nearest-even
__device__ __forceinline__ unsigned short f2bf(float f) {
    unsigned u = __float_as_uint(f);
    unsigned r = u + 0x7FFFu + ((u >> 16) & 1u);
    return (unsigned short)(r >> 16);
}

union FragU { uint4 q[2]; v16bf v; };

// Load one 16x32 bf16 WMMA operand fragment for this lane from an LDS tile.
// ISA layout (CDNA5 ISA, 16-bit A 16x32): lanes 0-15 hold K 0-7 and 16-23,
// lanes 16-31 hold K 8-15 and 24-31, for row/col = lane%16. B mirrors with N.
__device__ __forceinline__ v16bf ldsFrag(const unsigned short* lds, int row, int laneHi) {
    int base = row * LDS_STR + (laneHi ? 8 : 0);
    FragU u;
    u.q[0] = *reinterpret_cast<const uint4*>(lds + base);        // K 0-7   (or 8-15)
    u.q[1] = *reinterpret_cast<const uint4*>(lds + base + 16);   // K 16-23 (or 24-31)
    return u.v;
}

// CDNA5 async global->LDS copy: each lane moves 16B; tracked by ASYNCcnt.
__device__ __forceinline__ void async_ld16(void* ldsPtr, const void* g) {
    unsigned off = (unsigned)(uintptr_t)ldsPtr;   // low 32 bits of generic = LDS byte offset
    asm volatile("global_load_async_to_lds_b128 %0, %1, off" :: "v"(off), "v"(g) : "memory");
}
__device__ __forceinline__ void wait_async0() {
    asm volatile("s_wait_asynccnt 0x0" ::: "memory");
}

// ---------------------------------------------------------------- utilities
__global__ void k_zero(float* __restrict__ p, long long n) {
    long long i = (long long)blockIdx.x * blockDim.x + threadIdx.x;
    long long stride = (long long)gridDim.x * blockDim.x;
    for (; i < n; i += stride) p[i] = 0.0f;
}

__global__ void k_deg(const long long* __restrict__ ei, float* __restrict__ deg) {
    int e = blockIdx.x * blockDim.x + threadIdx.x;
    if (e < N_EDGES) atomicAdd(&deg[(int)ei[N_EDGES + e]], 1.0f);
}

__global__ void k_recip(float* __restrict__ deg) {
    int i = blockIdx.x * blockDim.x + threadIdx.x;
    if (i < N_NODES) {
        float d = deg[i];
        deg[i] = (d > 0.0f) ? (1.0f / d) : 0.0f;
    }
}

// one block per edge: scatter-add a full feature row (coalesced per block)
__global__ void k_scatter(const float* __restrict__ feat, const long long* __restrict__ ei,
                          float* __restrict__ aggr, int K) {
    int e = blockIdx.x;
    int s = (int)ei[e];
    int d = (int)ei[N_EDGES + e];
    const float* fs = feat + (size_t)s * K;
    float*       ad = aggr + (size_t)d * K;
    if (threadIdx.x == 0 && e + 1 < N_EDGES) {
        __builtin_prefetch(feat + (size_t)((int)ei[e + 1]) * K, 0, 0);  // global_prefetch
    }
    for (int c = threadIdx.x; c < K; c += blockDim.x)
        atomicAdd(&ad[c], fs[c]);
}

// one block per row: f32 -> bf16 (optionally scaled by rowScale[row]), x4 vectorized
__global__ void k_row_convert(const float* __restrict__ src, const float* __restrict__ scale,
                              unsigned short* __restrict__ dst, int K) {
    int r = blockIdx.x;
    float s = scale ? scale[r] : 1.0f;
    const float* p = src + (size_t)r * K;
    unsigned short* q = dst + (size_t)r * K;
    for (int c = threadIdx.x * 4; c < K; c += blockDim.x * 4) {
        float4 v = *reinterpret_cast<const float4*>(p + c);
        uint2 o;
        o.x = (unsigned)f2bf(v.x * s) | ((unsigned)f2bf(v.y * s) << 16);
        o.y = (unsigned)f2bf(v.z * s) | ((unsigned)f2bf(v.w * s) << 16);
        *reinterpret_cast<uint2*>(q + c) = o;
    }
}

// -------------------------------------------------- fused SAGE GEMM (WMMA)
// H = act( A @ Wl^T + X @ Wr^T + bias ), all operands pre-converted bf16:
// A (=deg_inv-scaled aggregation), X: [M,K]; Wl,Wr: [N,K] row-major.
// K % 32 == 0, N % 64 == 0. Double-buffered LDS staged via async global->LDS
// (ASYNCcnt), one workgroup barrier per K-step; f32 accumulation in WMMA.
__launch_bounds__(256)
__global__ void k_sage_gemm(const unsigned short* __restrict__ A, const unsigned short* __restrict__ X,
                            const unsigned short* __restrict__ Wl, const unsigned short* __restrict__ Wr,
                            const float* __restrict__ bias, float* __restrict__ H,
                            int M, int N, int K, int doRelu) {
    __shared__ unsigned short smem[2][4][64 * LDS_STR];   // [buf][A,X,Wl,Wr][tile]

    const int tid   = threadIdx.x;
    const int rLoc  = tid >> 2;          // 0..63 : tile row this thread stages
    const int kLoc  = (tid & 3) * 8;     // 0,8,16,24 : 8 consecutive bf16 (16B chunk)
    const int mBase = blockIdx.x * 64;
    const int nBase = blockIdx.y * 64;

    const int wv     = tid >> 5;         // wave id 0..7 (wave32)
    const int lane   = tid & 31;
    const int ln     = lane & 15;
    const int laneHi = (lane >> 4) & 1;
    const int waveM  = (wv & 3) * 16;    // 4 waves along M
    const int waveN  = (wv >> 2) * 32;   // 2 waves along N, each owns 2 N-subtiles

    int aRow = mBase + rLoc; if (aRow > M - 1) aRow = M - 1;   // clamp OOB rows
    const int nRow = nBase + rLoc;                              // N % 64 == 0 -> in range
    const size_t aOff = (size_t)aRow * K + kLoc;
    const size_t nOff = (size_t)nRow * K + kLoc;
    const int ldsOff  = rLoc * LDS_STR + kLoc;

    v8f acc0 = {0.f, 0.f, 0.f, 0.f, 0.f, 0.f, 0.f, 0.f};
    v8f acc1 = acc0;

    // issue one staging round: 4 async copy instructions per wave (A,X,Wl,Wr)
    auto issue_stage = [&](int kt, int b) {
        int k0 = kt * 32;
        async_ld16(&smem[b][0][ldsOff], A  + aOff + k0);
        async_ld16(&smem[b][1][ldsOff], X  + aOff + k0);
        async_ld16(&smem[b][2][ldsOff], Wl + nOff + k0);
        async_ld16(&smem[b][3][ldsOff], Wr + nOff + k0);
    };

    const int nK = K >> 5;
    issue_stage(0, 0);

    for (int kt = 0; kt < nK; ++kt) {
        wait_async0();        // my wave's stage-kt copies landed in LDS
        __syncthreads();      // every wave's stage-kt landed; compute(kt-1) done everywhere
        if (kt + 1 < nK) issue_stage(kt + 1, (kt + 1) & 1);   // overlap DMA with compute
        const int b = kt & 1;
        const unsigned short* bA  = smem[b][0];
        const unsigned short* bX  = smem[b][1];
        const unsigned short* bWl = smem[b][2];
        const unsigned short* bWr = smem[b][3];

        v16bf af = ldsFrag(bA,  waveM + ln, laneHi);
        v16bf xf = ldsFrag(bX,  waveM + ln, laneHi);
        v16bf l0 = ldsFrag(bWl, waveN + ln,      laneHi);
        v16bf l1 = ldsFrag(bWl, waveN + 16 + ln, laneHi);
        v16bf r0 = ldsFrag(bWr, waveN + ln,      laneHi);
        v16bf r1 = ldsFrag(bWr, waveN + 16 + ln, laneHi);

        acc0 = __builtin_amdgcn_wmma_f32_16x16x32_bf16(false, af, false, l0, (short)0, acc0, false, false);
        acc0 = __builtin_amdgcn_wmma_f32_16x16x32_bf16(false, xf, false, r0, (short)0, acc0, false, false);
        acc1 = __builtin_amdgcn_wmma_f32_16x16x32_bf16(false, af, false, l1, (short)0, acc1, false, false);
        acc1 = __builtin_amdgcn_wmma_f32_16x16x32_bf16(false, xf, false, r1, (short)0, acc1, false, false);
    }

    // ---- epilogue: bias + optional ReLU; C layout: VGPR r <-> M = r (+8 hi)
    int col0 = nBase + waveN + ln;
    int col1 = col0 + 16;
    float b0 = bias[col0];
    float b1 = bias[col1];
    #pragma unroll
    for (int r = 0; r < 8; ++r) {
        int row = mBase + waveM + r + (laneHi ? 8 : 0);
        if (row < M) {
            float v0 = acc0[r] + b0;
            float v1 = acc1[r] + b1;
            if (doRelu) { v0 = fmaxf(v0, 0.0f); v1 = fmaxf(v1, 0.0f); }
            H[(size_t)row * N + col0] = v0;
            H[(size_t)row * N + col1] = v1;
        }
    }
}

// ---------------------------------------------------------------- pooling
__global__ void k_pool(const float* __restrict__ h2, const long long* __restrict__ batch,
                       float* __restrict__ pooled, float* __restrict__ cnt) {
    int n = blockIdx.x;
    int g = (int)batch[n];
    atomicAdd(&pooled[g * D_H2 + threadIdx.x], h2[(size_t)n * D_H2 + threadIdx.x]);
    if (threadIdx.x == 0) atomicAdd(&cnt[g], 1.0f);
}

// ------------------------------------------------------------- MLP head
__global__ void k_head(const float* __restrict__ pooled, const float* __restrict__ cnt,
                       const float* __restrict__ W3, const float* __restrict__ b3,
                       const float* __restrict__ W4, const float* __restrict__ b4,
                       float* __restrict__ out) {
    int g = threadIdx.x;                 // 256 threads, one graph each
    if (g >= N_GRAPHS) return;
    float inv = 1.0f / fmaxf(cnt[g], 1.0f);
    float p[D_H2];
    #pragma unroll 4
    for (int c = 0; c < D_H2; ++c) p[c] = pooled[g * D_H2 + c] * inv;
    float h3[D_H3];
    for (int j = 0; j < D_H3; ++j) {
        float acc = b3[j];
        const float* w = W3 + j * D_H2;
        #pragma unroll 4
        for (int c = 0; c < D_H2; ++c) acc = fmaf(p[c], w[c], acc);
        h3[j] = acc;
    }
    float o0 = b4[0], o1 = b4[1];
    #pragma unroll 4
    for (int j = 0; j < D_H3; ++j) {
        o0 = fmaf(h3[j], W4[j], o0);
        o1 = fmaf(h3[j], W4[D_H3 + j], o1);
    }
    float m  = fmaxf(o0, o1);
    float e0 = __expf(o0 - m), e1 = __expf(o1 - m);
    float s  = e0 + e1;
    out[g * 2 + 0] = e0 / s;
    out[g * 2 + 1] = e1 / s;
}

// ---------------------------------------------------------------- launch
extern "C" void kernel_launch(void* const* d_in, const int* in_sizes, int n_in,
                              void* d_out, int out_size, void* d_ws, size_t ws_size,
                              hipStream_t stream) {
    (void)in_sizes; (void)n_in; (void)out_size; (void)ws_size;
    const float*     x     = (const float*)d_in[0];
    const long long* ei    = (const long long*)d_in[1];
    const long long* batch = (const long long*)d_in[2];
    const float* W1l = (const float*)d_in[3];
    const float* b1l = (const float*)d_in[4];
    const float* W1r = (const float*)d_in[5];
    const float* W2l = (const float*)d_in[6];
    const float* b2l = (const float*)d_in[7];
    const float* W2r = (const float*)d_in[8];
    const float* W3  = (const float*)d_in[9];
    const float* b3  = (const float*)d_in[10];
    const float* W4  = (const float*)d_in[11];
    const float* b4  = (const float*)d_in[12];
    float* out = (float*)d_out;

    char*  ws  = (char*)d_ws;
    size_t off = 0;
    auto alloc = [&](size_t bytes) -> void* {
        char* p = ws + off;
        off = (off + bytes + 255) & ~(size_t)255;
        return p;
    };
    float* deg    = (float*)alloc(sizeof(float) * N_NODES);                    // -> deg_inv
    float* aggr   = (float*)alloc(sizeof(float) * (size_t)N_NODES * D_IN);     // reused L2
    float* h1     = (float*)alloc(sizeof(float) * (size_t)N_NODES * D_H1);
    float* h2     = (float*)alloc(sizeof(float) * (size_t)N_NODES * D_H2);
    unsigned short* xb   = (unsigned short*)alloc(sizeof(short) * (size_t)N_NODES * D_IN);
    unsigned short* ab   = (unsigned short*)alloc(sizeof(short) * (size_t)N_NODES * D_IN); // reused L2
    unsigned short* h1b  = (unsigned short*)alloc(sizeof(short) * (size_t)N_NODES * D_H1);
    unsigned short* w1lb = (unsigned short*)alloc(sizeof(short) * D_H1 * D_IN);
    unsigned short* w1rb = (unsigned short*)alloc(sizeof(short) * D_H1 * D_IN);
    unsigned short* w2lb = (unsigned short*)alloc(sizeof(short) * D_H2 * D_H1);
    unsigned short* w2rb = (unsigned short*)alloc(sizeof(short) * D_H2 * D_H1);
    float* pooled = (float*)alloc(sizeof(float) * N_GRAPHS * D_H2);
    float* cnt    = (float*)alloc(sizeof(float) * N_GRAPHS);

    // degree + inverse
    k_zero  <<<512, 256, 0, stream>>>(deg, (long long)N_NODES);
    k_deg   <<<(N_EDGES + 255) / 256, 256, 0, stream>>>(ei, deg);
    k_recip <<<(N_NODES + 255) / 256, 256, 0, stream>>>(deg);

    // one-time bf16 mirrors of weights and x
    k_row_convert<<<D_H1, 256, 0, stream>>>(W1l, nullptr, w1lb, D_IN);
    k_row_convert<<<D_H1, 256, 0, stream>>>(W1r, nullptr, w1rb, D_IN);
    k_row_convert<<<D_H2, 256, 0, stream>>>(W2l, nullptr, w2lb, D_H1);
    k_row_convert<<<D_H2, 256, 0, stream>>>(W2r, nullptr, w2rb, D_H1);
    k_row_convert<<<N_NODES, 256, 0, stream>>>(x, nullptr, xb, D_IN);

    // layer 1: mean-aggregate x (f32 atomics), fold deg_inv into bf16 convert,
    // then async-staged WMMA GEMM + ReLU
    k_zero       <<<4096, 256, 0, stream>>>(aggr, (long long)N_NODES * D_IN);
    k_scatter    <<<N_EDGES, 256, 0, stream>>>(x, ei, aggr, D_IN);
    k_row_convert<<<N_NODES, 256, 0, stream>>>(aggr, deg, ab, D_IN);
    dim3 g1((N_NODES + 63) / 64, D_H1 / 64);
    k_sage_gemm<<<g1, 256, 0, stream>>>(ab, xb, w1lb, w1rb, b1l, h1,
                                        N_NODES, D_H1, D_IN, 1);

    // layer 2: mean-aggregate h1 (reuse aggr buffer), convert, GEMM + ReLU
    k_zero       <<<4096, 256, 0, stream>>>(aggr, (long long)N_NODES * D_H1);
    k_scatter    <<<N_EDGES, 256, 0, stream>>>(h1, ei, aggr, D_H1);
    k_row_convert<<<N_NODES, 256, 0, stream>>>(aggr, deg, ab, D_H1);   // reuse ab
    k_row_convert<<<N_NODES, 256, 0, stream>>>(h1, nullptr, h1b, D_H1);
    dim3 g2((N_NODES + 63) / 64, D_H2 / 64);
    k_sage_gemm<<<g2, 256, 0, stream>>>(ab, h1b, w2lb, w2rb, b2l, h2,
                                        N_NODES, D_H2, D_H1, 1);

    // global mean pool + MLP head + softmax
    k_zero <<<64, 256, 0, stream>>>(pooled, (long long)(N_GRAPHS * D_H2));
    k_zero <<<1, 256, 0, stream>>>(cnt, (long long)N_GRAPHS);
    k_pool <<<N_NODES, D_H2, 0, stream>>>(h2, batch, pooled, cnt);
    k_head <<<1, 256, 0, stream>>>(pooled, cnt, W3, b3, W4, b4, out);
}
